// SingleHeadSelfAttention_1279900254316
// MI455X (gfx1250) — matrix operand order repaired and checked by reading, not compiled, not run on hardware
//
#include <hip/hip_runtime.h>
#include <hip/hip_bf16.h>

// ---------------------------------------------------------------------------
// Single-head causal self-attention for MI455X (gfx1250), wave32 + WMMA bf16.
// B=4, T=4096, d_model=1024, d_head=128. fp32 in/out, bf16 tensor-core math.
// Flash attention with optional split-K(keys)=2 + LSE merge (gated on ws_size).
// ---------------------------------------------------------------------------

typedef __bf16 bf16;
typedef bf16  v8bf  __attribute__((ext_vector_type(8)));
typedef bf16  v16bf __attribute__((ext_vector_type(16)));
typedef float v8f   __attribute__((ext_vector_type(8)));

#define BATCH   4
#define SEQ     4096
#define DMODEL  1024
#define DHEAD   128
#define BT      (BATCH * SEQ)

static __device__ __forceinline__ v16bf cat8(v8bf lo, v8bf hi) {
    return __builtin_shufflevector(lo, hi, 0,1,2,3,4,5,6,7,8,9,10,11,12,13,14,15);
}

static __device__ __forceinline__ v8f wmma_bf16(v16bf a, v16bf b, v8f c) {
    // D = A(16x32) * B(32x16) + C(16x16 f32)
    return __builtin_amdgcn_wmma_f32_16x16x32_bf16(false, a, false, b,
                                                   (short)0, c, false, false);
}

// ---------------------------------------------------------------------------
// Kernel 0: convert weights f32 [1024][128] -> bf16 transposed [3][128][1024]
// ---------------------------------------------------------------------------
__global__ void wconv_kernel(const float* __restrict__ Wq,
                             const float* __restrict__ Wk,
                             const float* __restrict__ Wv,
                             bf16* __restrict__ Wt) {
    int idx = blockIdx.x * 256 + threadIdx.x;          // 0 .. 3*1024*128-1
    int w   = idx >> 17;                               // /131072
    int rem = idx & 131071;
    int k   = rem >> 7;                                // d_model index
    int n   = rem & 127;                               // head-dim index
    const float* W = (w == 0) ? Wq : ((w == 1) ? Wk : Wv);
    Wt[(size_t)w * (DHEAD * DMODEL) + (size_t)n * DMODEL + k] = (bf16)W[k * DHEAD + n];
}

// ---------------------------------------------------------------------------
// Kernel 1: QKV projection. One wave computes a 16-row x 128-col tile of one
// of {Q, K, V} via WMMA bf16 (K=1024 in 32 chunks of 32).
//   Q, K stored [B][T][128] bf16 ; V stored transposed [B][128][T] bf16.
// Explicit double buffering: chunk c+1's A/B fragments are in flight while
// chunk c's 8 WMMAs execute (uniform control flow, clamped last iteration).
// ---------------------------------------------------------------------------
__global__ __launch_bounds__(128) void qkv_proj_kernel(
        const float* __restrict__ x,     // [B][T][1024] f32
        const bf16*  __restrict__ Wt,    // [3][128][1024] bf16 (transposed)
        bf16* __restrict__ Q,            // [B][T][128]
        bf16* __restrict__ K,            // [B][T][128]
        bf16* __restrict__ Vt)           // [B][128][T]
{
    const int lane = threadIdx.x & 31;
    const int warp = threadIdx.x >> 5;
    const int w    = blockIdx.x * 4 + warp;   // 0 .. 3071
    const int tgt  = w / 1024;                // 0=Q 1=K 2=V
    const int rem  = w - tgt * 1024;
    const int b    = rem >> 8;                // batch
    const int q0   = (rem & 255) << 4;        // first token of tile
    const int nl   = lane & 15;
    const int g    = lane >> 4;

    const bf16*  Wtp  = Wt + (size_t)tgt * (DHEAD * DMODEL);
    const float* xrow = x + ((size_t)(b * SEQ + q0 + nl)) * DMODEL;
    const bf16*  wrow = Wtp + (size_t)nl * DMODEL;

    v8f acc[8] = {};

    // Prologue: loads for chunk 0.
    v8f   xa = *(const v8f*)(xrow + g * 16);
    v8f   xb = *(const v8f*)(xrow + g * 16 + 8);
    v16bf bcur[8];
    #pragma unroll
    for (int t = 0; t < 8; ++t)
        bcur[t] = *(const v16bf*)(wrow + (size_t)(t * 16) * DMODEL + g * 16);

    for (int c = 0; c < 32; ++c) {
        // Issue next chunk's loads first (clamped: last iter reloads chunk 31).
        const int cn  = (c < 31) ? (c + 1) : 31;
        const int k0n = cn * 32 + g * 16;
        const int kbn = cn * 32 + g * 16;
        __builtin_prefetch(xrow + k0n + 32, 0, 3);
        v8f xa2 = *(const v8f*)(xrow + k0n);
        v8f xb2 = *(const v8f*)(xrow + k0n + 8);
        v16bf bnext[8];
        #pragma unroll
        for (int t = 0; t < 8; ++t)
            bnext[t] = *(const v16bf*)(wrow + (size_t)(t * 16) * DMODEL + kbn);

        // Compute with current chunk.
        v16bf afrag = cat8(__builtin_convertvector(xa, v8bf),
                           __builtin_convertvector(xb, v8bf));
        #pragma unroll
        for (int t = 0; t < 8; ++t)
            acc[t] = wmma_bf16(afrag, bcur[t], acc[t]);

        xa = xa2; xb = xb2;
        #pragma unroll
        for (int t = 0; t < 8; ++t) bcur[t] = bnext[t];
    }

    // Epilogue: C layout -> lane l, vgpr r holds (M = r + 8*g, N = nl)
    if (tgt == 2) {
        #pragma unroll
        for (int t = 0; t < 8; ++t)
            #pragma unroll
            for (int r = 0; r < 8; ++r) {
                int tok = q0 + r + 8 * g;
                int h   = t * 16 + nl;
                Vt[((size_t)(b * DHEAD + h)) * SEQ + tok] = (bf16)acc[t][r];
            }
    } else {
        bf16* dst = (tgt == 0) ? Q : K;
        #pragma unroll
        for (int t = 0; t < 8; ++t)
            #pragma unroll
            for (int r = 0; r < 8; ++r) {
                int tok = q0 + r + 8 * g;
                int h   = t * 16 + nl;
                dst[((size_t)(b * SEQ + tok)) * DHEAD + h] = (bf16)acc[t][r];
            }
    }
}

// ---------------------------------------------------------------------------
// Kernel 2: causal flash attention, optional interleaved split over key blocks.
// One wave owns 16 query rows and key blocks {s, s+nsplit, ...} of 32 keys.
// S = Q*K^T (8 WMMAs), online softmax (f32 butterfly shfl in 16-lane groups),
// P staged via LDS (C-layout -> A-layout), O += P*V (8 WMMAs, transposed V).
// nsplit==1: write normalized fp32 out. nsplit==2: write partial O + (m,l).
// ---------------------------------------------------------------------------
__global__ __launch_bounds__(128) void attn_kernel(
        const bf16* __restrict__ Q,      // [B][T][128]
        const bf16* __restrict__ K,      // [B][T][128]
        const bf16* __restrict__ Vt,     // [B][128][T]
        float* __restrict__ out,         // [B][T][128] f32 (nsplit==1)
        float* __restrict__ Opart,       // [nsplit][B*T][128] f32 (nsplit==2)
        float* __restrict__ Ml,          // [nsplit][B*T][2] f32  (nsplit==2)
        int nsplit, int ns_log2)
{
    __shared__ __align__(16) bf16 plds[4 * 16 * 32];   // 1KB per wave

    const int lane = threadIdx.x & 31;
    const int warp = threadIdx.x >> 5;
    bf16* pbase = plds + warp * (16 * 32);

    const int w    = blockIdx.x * 4 + warp;  // 0 .. nsplit*1024-1
    const int s    = w & (nsplit - 1);
    const int tile = w >> ns_log2;           // 0 .. 1023
    const int b    = tile >> 8;
    const int q0   = (tile & 255) << 4;
    const int nl   = lane & 15;
    const int g    = lane >> 4;

    // Q fragments (A layout), kept resident: 4 chunks of K=32
    const bf16* qrow = Q + ((size_t)(b * SEQ + q0 + nl)) * DHEAD;
    v16bf qf[4];
    #pragma unroll
    for (int c = 0; c < 4; ++c) {
        int k0 = c * 32 + g * 16;
        qf[c] = cat8(*(const v8bf*)(qrow + k0), *(const v8bf*)(qrow + k0 + 8));
    }

    v8f o[8] = {};
    float mprev[8], lsum[8], arow[8];
    #pragma unroll
    for (int r = 0; r < 8; ++r) { mprev[r] = -INFINITY; lsum[r] = 0.0f; }

    const float scale = 0.08838834764831845f;   // 1/sqrt(128)
    const int npairs = (q0 + 16 + 31) >> 5;     // key blocks of 32
    const int step32 = nsplit << 5;             // key stride in tokens

    for (int jp = s; jp < npairs; jp += nsplit) {
        const int kb = jp << 5;

        // Prefetch next key block of this wave toward the WGP caches.
        if (kb + step32 < q0 + 16) {
            __builtin_prefetch(K + ((size_t)(b * SEQ + kb + step32 + nl)) * DHEAD + g * 64, 0, 3);
            __builtin_prefetch(Vt + ((size_t)(b * DHEAD + lane * 4)) * SEQ + kb + step32, 0, 3);
        }

        // --- S = Q * K^T : two 16x16 tiles. Batch loads (4 frags) per 2 chunks.
        v8f s0 = {}, s1 = {};
        {
            const bf16* krow0 = K + ((size_t)(b * SEQ + kb + nl)) * DHEAD;
            const bf16* krow1 = krow0 + (size_t)16 * DHEAD;
            #pragma unroll
            for (int cc = 0; cc < 2; ++cc) {
                v16bf kf[4];
                #pragma unroll
                for (int c2 = 0; c2 < 2; ++c2) {
                    int kb0 = (cc * 2 + c2) * 32 + g * 16;
                    kf[c2 * 2 + 0] = *(const v16bf*)(krow0 + kb0);
                    kf[c2 * 2 + 1] = *(const v16bf*)(krow1 + kb0);
                }
                #pragma unroll
                for (int c2 = 0; c2 < 2; ++c2) {
                    s0 = wmma_bf16(qf[cc * 2 + c2], kf[c2 * 2 + 0], s0);
                    s1 = wmma_bf16(qf[cc * 2 + c2], kf[c2 * 2 + 1], s1);
                }
            }
        }

        // --- scale + causal mask + online softmax update ---
        #pragma unroll
        for (int r = 0; r < 8; ++r) {
            int m  = r + 8 * g;
            int qg = q0 + m;
            float e0 = (kb + nl      <= qg) ? s0[r] * scale : -INFINITY;
            float e1 = (kb + 16 + nl <= qg) ? s1[r] * scale : -INFINITY;
            float x = fmaxf(e0, e1);
            x = fmaxf(x, __shfl_xor(x, 1));
            x = fmaxf(x, __shfl_xor(x, 2));
            x = fmaxf(x, __shfl_xor(x, 4));
            x = fmaxf(x, __shfl_xor(x, 8));
            float mnew = fmaxf(mprev[r], x);
            float p0 = __expf(e0 - mnew);
            float p1 = __expf(e1 - mnew);
            float rs = p0 + p1;
            rs += __shfl_xor(rs, 1);
            rs += __shfl_xor(rs, 2);
            rs += __shfl_xor(rs, 4);
            rs += __shfl_xor(rs, 8);
            float alpha = __expf(mprev[r] - mnew);
            lsum[r]  = lsum[r] * alpha + rs;
            mprev[r] = mnew;
            arow[r]  = alpha;
            // stage P (C layout element (m, t*16+nl)) into LDS as bf16
            pbase[m * 32 + nl]      = (bf16)p0;
            pbase[m * 32 + 16 + nl] = (bf16)p1;
        }

        // --- O += P(16x32) * V(32x16 per head tile) ---
        // Issue all independent V global loads first, then the LDS P read
        // (its ds wait overlaps the global latency), then rescale + 8 WMMAs.
        {
            const bf16* vbase = Vt + ((size_t)(b * DHEAD + nl)) * SEQ + kb + g * 16;
            v16bf vf[8];
            #pragma unroll
            for (int t = 0; t < 8; ++t)
                vf[t] = *(const v16bf*)(vbase + (size_t)(t * 16) * SEQ);

            const bf16* pr = pbase + nl * 32 + g * 16;          // A layout
            v16bf pf = cat8(*(const v8bf*)(pr), *(const v8bf*)(pr + 8));

            #pragma unroll
            for (int t = 0; t < 8; ++t)
                #pragma unroll
                for (int r = 0; r < 8; ++r)
                    o[t][r] *= arow[r];

            #pragma unroll
            for (int t = 0; t < 8; ++t)
                o[t] = wmma_bf16(pf, vf[t], o[t]);
        }
    }

    // --- epilogue ---
    if (nsplit == 1) {
        #pragma unroll
        for (int r = 0; r < 8; ++r) arow[r] = 1.0f / lsum[r];
        #pragma unroll
        for (int t = 0; t < 8; ++t)
            #pragma unroll
            for (int r = 0; r < 8; ++r) {
                int m = r + 8 * g;
                out[((size_t)(b * SEQ + q0 + m)) * DHEAD + t * 16 + nl] = o[t][r] * arow[r];
            }
    } else {
        float* op = Opart + (size_t)s * BT * DHEAD;
        #pragma unroll
        for (int t = 0; t < 8; ++t)
            #pragma unroll
            for (int r = 0; r < 8; ++r) {
                int m = r + 8 * g;
                op[((size_t)(b * SEQ + q0 + m)) * DHEAD + t * 16 + nl] = o[t][r];
            }
        if (nl == 0) {
            float* mlp = Ml + (size_t)s * BT * 2;
            #pragma unroll
            for (int r = 0; r < 8; ++r) {
                size_t row = (size_t)(b * SEQ + q0 + r + 8 * g);
                mlp[row * 2 + 0] = mprev[r];
                mlp[row * 2 + 1] = lsum[r];
            }
        }
    }
}

// ---------------------------------------------------------------------------
// Kernel 3: merge the two split-K partials with a log-sum-exp combine.
// ---------------------------------------------------------------------------
__global__ void merge_kernel(const float* __restrict__ Opart,
                             const float* __restrict__ Ml,
                             float* __restrict__ out) {
    int idx = blockIdx.x * 256 + threadIdx.x;    // 0 .. B*T*128-1
    int row = idx >> 7;
    float m0 = Ml[(size_t)row * 2 + 0];
    float l0 = Ml[(size_t)row * 2 + 1];
    float m1 = Ml[(size_t)BT * 2 + (size_t)row * 2 + 0];
    float l1 = Ml[(size_t)BT * 2 + (size_t)row * 2 + 1];
    float m  = fmaxf(m0, m1);                    // finite: split 0 always runs
    float w0 = __expf(m0 - m);
    float w1 = __expf(m1 - m);                   // 0 if m1 == -inf
    float num = w0 * Opart[idx] + w1 * Opart[(size_t)BT * DHEAD + idx];
    out[idx] = num / (w0 * l0 + w1 * l1);
}

// ---------------------------------------------------------------------------
// Host launcher
// ---------------------------------------------------------------------------
extern "C" void kernel_launch(void* const* d_in, const int* in_sizes, int n_in,
                              void* d_out, int out_size, void* d_ws, size_t ws_size,
                              hipStream_t stream) {
    const float* x  = (const float*)d_in[0];
    const float* Wq = (const float*)d_in[1];
    const float* Wk = (const float*)d_in[2];
    const float* Wv = (const float*)d_in[3];
    float* out = (float*)d_out;

    // Workspace layout (bytes):
    //   [0, 786432)        Wt    : 3 x 128 x 1024 bf16 (transposed weights)
    //   [786432,  +4MB)    Q     : 4 x 4096 x 128 bf16
    //   [.., +4MB)         K     : 4 x 4096 x 128 bf16
    //   [.., +4MB)         Vt    : 4 x 128 x 4096 bf16
    //   [13369344, +16MB)  Opart : 2 x (B*T) x 128 f32   (split only)
    //   [30146560, +256KB) Ml    : 2 x (B*T) x 2 f32     (split only)
    char* ws = (char*)d_ws;
    bf16*  Wt    = (bf16*)ws;
    bf16*  Q     = (bf16*)(ws + 786432);
    bf16*  K     = (bf16*)(ws + 786432 + 4194304);
    bf16*  Vt    = (bf16*)(ws + 786432 + 8388608);
    float* Opart = (float*)(ws + 13369344);
    float* Ml    = (float*)(ws + 30146560);
    const size_t NEED_SPLIT = 30408704;

    const int nsplit  = (ws_size >= NEED_SPLIT) ? 2 : 1;
    const int ns_log2 = (nsplit == 2) ? 1 : 0;

    // 0) weight convert+transpose: 393216 elements
    wconv_kernel<<<1536, 256, 0, stream>>>(Wq, Wk, Wv, Wt);

    // 1) QKV projection: 3 targets x 4 batches x 256 tiles = 3072 waves
    qkv_proj_kernel<<<768, 128, 0, stream>>>(x, Wt, Q, K, Vt);

    // 2) causal flash attention: nsplit x 1024 waves, 4 waves/block
    attn_kernel<<<nsplit * 256, 128, 0, stream>>>(Q, K, Vt, out, Opart, Ml,
                                                  nsplit, ns_log2);

    // 3) merge partials (split path only)
    if (nsplit == 2)
        merge_kernel<<<(BT * DHEAD) / 256, 256, 0, stream>>>(Opart, Ml, out);
}
